// PointNetSAMsg_49082886258877
// MI455X (gfx1250) — matrix operand order repaired and checked by reading, not compiled, not run on hardware
//
#include <hip/hip_runtime.h>

typedef __attribute__((ext_vector_type(16))) _Float16 v16h;
typedef __attribute__((ext_vector_type(8)))  _Float16 v8h;
typedef __attribute__((ext_vector_type(8)))  float    v8f;

#define NPTS  16384
#define NSAMP 1024
#define BATCH 8
#define EPSF  1e-5f

// ---------------------------------------------------------------------------
// Farthest point sampling: one 1024-thread block per batch. Coords cached in
// registers (16 points/thread), LDS argmax tree, first-index tie-break.
// Also emits new_xyz (B,S,3) scratch copy and the (B,3,S) slice of d_out.
// ---------------------------------------------------------------------------
__global__ __launch_bounds__(1024)
void fps_kernel(const float* __restrict__ xyz, int* __restrict__ fps_idx,
                float* __restrict__ nxyz, float* __restrict__ out_xyz) {
  const int b = blockIdx.x;
  const int t = threadIdx.x;
  const int PER = NPTS / 1024;
  __shared__ float sval[1024];
  __shared__ int   sidx[1024];
  __shared__ int   sh_fps[1024];
  __shared__ float scx, scy, scz;
  const float* X = xyz + (size_t)b * 3 * NPTS;
  float px[PER], py[PER], pz[PER], dist[PER];
  const int base = t * PER;
  for (int i = 0; i < PER; ++i) {
    px[i] = X[base + i];
    py[i] = X[NPTS + base + i];
    pz[i] = X[2 * NPTS + base + i];
    dist[i] = 1e10f;
  }
  int farthest = 0;
  for (int it = 0; it < NSAMP; ++it) {
    if (t == 0) {
      sh_fps[it] = farthest;
      fps_idx[b * NSAMP + it] = farthest;
      scx = X[farthest]; scy = X[NPTS + farthest]; scz = X[2 * NPTS + farthest];
    }
    __syncthreads();
    float cx = scx, cy = scy, cz = scz;
    float best = -1.0f; int bidx = base;
    for (int i = 0; i < PER; ++i) {
      float dx = px[i] - cx, dy = py[i] - cy, dz = pz[i] - cz;
      float d = dx * dx + dy * dy + dz * dz;
      float nd = dist[i] = fminf(dist[i], d);
      if (nd > best) { best = nd; bidx = base + i; }
    }
    sval[t] = best; sidx[t] = bidx;
    __syncthreads();
    for (int off = 512; off > 0; off >>= 1) {
      if (t < off) {
        float ov = sval[t + off]; int oi = sidx[t + off];
        if (ov > sval[t] || (ov == sval[t] && oi < sidx[t])) {
          sval[t] = ov; sidx[t] = oi;
        }
      }
      __syncthreads();
    }
    farthest = sidx[0];
  }
  __syncthreads();
  {
    int idx = sh_fps[t];
    float x = X[idx], y = X[NPTS + idx], z = X[2 * NPTS + idx];
    size_t q = (size_t)b * NSAMP + t;
    nxyz[q * 3 + 0] = x; nxyz[q * 3 + 1] = y; nxyz[q * 3 + 2] = z;
    out_xyz[(size_t)b * 3 * NSAMP + 0 * NSAMP + t] = x;
    out_xyz[(size_t)b * 3 * NSAMP + 1 * NSAMP + t] = y;
    out_xyz[(size_t)b * 3 * NSAMP + 2 * NSAMP + t] = z;
  }
}

// ---------------------------------------------------------------------------
// Ball query: one wave32 per query point; scan points in ascending index
// order in chunks of 32, ballot+prefix-popcount to write the first NS hits
// (== reference's sort-by-index take-first). Pad tail with first hit.
// ---------------------------------------------------------------------------
template <int NS>
__global__ __launch_bounds__(256)
void ball_query_kernel(const float* __restrict__ xyz,
                       const float* __restrict__ nxyz,
                       int* __restrict__ gidx, float r2) {
  int q = blockIdx.x * 8 + (threadIdx.x >> 5);
  int lane = threadIdx.x & 31;
  int b = q >> 10;
  const float* X = xyz + (size_t)b * 3 * NPTS;
  float cx = nxyz[q * 3 + 0], cy = nxyz[q * 3 + 1], cz = nxyz[q * 3 + 2];
  int* out = gidx + (size_t)q * NS;
  int count = 0;
  for (int base = 0; base < NPTS && count < NS; base += 32) {
    int n = base + lane;
    float dx = X[n] - cx, dy = X[NPTS + n] - cy, dz = X[2 * NPTS + n] - cz;
    bool pred = (dx * dx + dy * dy + dz * dz) <= r2;
    unsigned mask = (unsigned)__ballot((int)pred);
    int pos = count + __popc(mask & ((1u << lane) - 1u));
    if (pred && pos < NS) out[pos] = n;
    count += __popc(mask);
  }
  if (count > NS) count = NS;
  int first = (count > 0) ? out[0] : 0;
  for (int j = count + lane; j < NS; j += 32) out[j] = first;
}

// ---------------------------------------------------------------------------
// Build f16 feature rows: [pts gather(3) | relative xyz(3) | zero pad] -> 32
// ---------------------------------------------------------------------------
template <int NS>
__global__ __launch_bounds__(256)
void build_features_kernel(const float* __restrict__ xyz,
                           const float* __restrict__ pts,
                           const float* __restrict__ nxyz,
                           const int* __restrict__ gidx,
                           _Float16* __restrict__ X0, int R) {
  int t = blockIdx.x * 256 + threadIdx.x;
  if (t >= R) return;
  int q = t / NS;
  int b = q >> 10;
  int g = gidx[t];
  g = g < 0 ? 0 : (g >= NPTS ? NPTS - 1 : g);
  const float* Pb = pts + (size_t)b * 3 * NPTS;
  const float* Xb = xyz + (size_t)b * 3 * NPTS;
  float f0 = Pb[g], f1 = Pb[NPTS + g], f2 = Pb[2 * NPTS + g];
  float f3 = Xb[g] - nxyz[q * 3 + 0];
  float f4 = Xb[NPTS + g] - nxyz[q * 3 + 1];
  float f5 = Xb[2 * NPTS + g] - nxyz[q * 3 + 2];
  v16h lo, hi;
  for (int i = 0; i < 16; ++i) { lo[i] = (_Float16)0.f; hi[i] = (_Float16)0.f; }
  lo[0] = (_Float16)f0; lo[1] = (_Float16)f1; lo[2] = (_Float16)f2;
  lo[3] = (_Float16)f3; lo[4] = (_Float16)f4; lo[5] = (_Float16)f5;
  v16h* dst = (v16h*)(X0 + (size_t)t * 32);
  dst[0] = lo;
  dst[1] = hi;
}

// ---------------------------------------------------------------------------
// Weight prep: f32 [CO x CI] -> f16 [CO x KP] zero-padded along K.
// ---------------------------------------------------------------------------
__global__ __launch_bounds__(256)
void prep_weight_kernel(const float* __restrict__ W, _Float16* __restrict__ Wf,
                        int CO, int CI, int KP) {
  int t = blockIdx.x * 256 + threadIdx.x;
  if (t >= CO * KP) return;
  int o = t / KP, k = t % KP;
  Wf[t] = (_Float16)((k < CI) ? W[o * CI + k] : 0.f);
}

__global__ void zero_stats_kernel(float* __restrict__ s) { s[threadIdx.x] = 0.f; }

// ---------------------------------------------------------------------------
// WMMA GEMM  Y[R x CO] = X[R x KP] * W^T + bias, with per-channel sum/sumsq
// (for batch-global BN) via LDS then global atomics. 8 waves per block, each
// wave owns a 16-row tile; loops over 16-wide output-channel tiles and K/32.
// A fragment: lane(m,half) holds K in [16*half, 16*half+16) (contiguous,
// matching B). B fragment: lane-half
// selects K range [16*half, 16*half+16), N = lane&15.
// ---------------------------------------------------------------------------
template <int KP, int CO>
__global__ __launch_bounds__(256)
void gemm_bn_kernel(const _Float16* __restrict__ X, const _Float16* __restrict__ Wf,
                    const float* __restrict__ bias, _Float16* __restrict__ Y,
                    float* __restrict__ gsum, float* __restrict__ gsumsq, int R) {
  __shared__ float s_sum[CO];
  __shared__ float s_sq[CO];
  const int tid = threadIdx.x;
  if (tid < CO) { s_sum[tid] = 0.f; s_sq[tid] = 0.f; }
  __syncthreads();
  const int wave = tid >> 5, lane = tid & 31;
  const int half = lane >> 4, l15 = lane & 15;
  const int row0 = (blockIdx.x * 8 + wave) * 16;
  const _Float16* xrow = X + (size_t)(row0 + l15) * KP;
  for (int nt = 0; nt < CO / 16; ++nt) {
    const int n = nt * 16 + l15;
    const _Float16* wrow = Wf + (size_t)n * KP + 16 * half;
    v8f acc = {};
    for (int ks = 0; ks < KP / 32; ++ks) {
      v8h alo = *(const v8h*)(xrow + ks * 32 + 16 * half);
      v8h ahi = *(const v8h*)(xrow + ks * 32 + 16 * half + 8);
      v16h a = __builtin_shufflevector(alo, ahi, 0, 1, 2, 3, 4, 5, 6, 7,
                                       8, 9, 10, 11, 12, 13, 14, 15);
      v16h bf = *(const v16h*)(wrow + ks * 32);
      acc = __builtin_amdgcn_wmma_f32_16x16x32_f16(
          false, a, false, bf, (short)0, acc, false, false);
    }
    float bv = bias[n];
    float s = 0.f, s2 = 0.f;
    _Float16 ov[8];
    for (int r = 0; r < 8; ++r) {
      float v = acc[r] + bv;
      s += v; s2 += v * v;
      ov[r] = (_Float16)v;
    }
    atomicAdd(&s_sum[n], s);
    atomicAdd(&s_sq[n], s2);
    for (int r = 0; r < 8; ++r)
      Y[(size_t)(row0 + r + 8 * half) * CO + n] = ov[r];
  }
  __syncthreads();
  if (tid < CO) {
    atomicAdd(&gsum[tid], s_sum[tid]);
    atomicAdd(&gsumsq[tid], s_sq[tid]);
  }
}

// ---------------------------------------------------------------------------
// In-place BN(training stats) + ReLU on f16 activations.
// ---------------------------------------------------------------------------
template <int CO>
__global__ __launch_bounds__(256)
void bn_relu_kernel(_Float16* __restrict__ X, const float* __restrict__ gsum,
                    const float* __restrict__ gsumsq,
                    const float* __restrict__ gamma,
                    const float* __restrict__ beta, int R) {
  size_t i = (size_t)blockIdx.x * 256 + threadIdx.x;
  if (i >= (size_t)R * CO) return;
  int c = (int)(i % CO);
  float cnt = (float)R;
  float mu = gsum[c] / cnt;
  float var = gsumsq[c] / cnt - mu * mu;
  float inv = rsqrtf(var + EPSF);
  float v = gamma[c] * ((float)X[i] - mu) * inv + beta[c];
  X[i] = (_Float16)fmaxf(v, 0.f);
}

// ---------------------------------------------------------------------------
// Final layer: BN + ReLU + max over K, writing (B, 320, S) slice of d_out.
// ---------------------------------------------------------------------------
template <int NS, int CO>
__global__ __launch_bounds__(256)
void bn_relu_max_kernel(const _Float16* __restrict__ Y,
                        const float* __restrict__ gsum,
                        const float* __restrict__ gsumsq,
                        const float* __restrict__ gamma,
                        const float* __restrict__ beta,
                        float* __restrict__ out, int c_base) {
  int t = blockIdx.x * 256 + threadIdx.x;
  if (t >= BATCH * NSAMP * CO) return;
  int c = t % CO;
  int q = t / CO;
  int b = q >> 10, s = q & 1023;
  float cnt = (float)(BATCH * NSAMP) * (float)NS;
  float mu = gsum[c] / cnt;
  float var = gsumsq[c] / cnt - mu * mu;
  float inv = rsqrtf(var + EPSF);
  float g = gamma[c], be = beta[c];
  float m = 0.f;  // ReLU output is >= 0, so 0 is a valid identity for max
  for (int k = 0; k < NS; ++k) {
    float y = (float)Y[((size_t)q * NS + k) * CO + c];
    float v = fmaxf(g * (y - mu) * inv + be, 0.f);
    m = fmaxf(m, v);
  }
  out[(size_t)b * (320 * NSAMP) + (size_t)(c_base + c) * NSAMP + s] = m;
}

// ---------------------------------------------------------------------------
// Host-side branch driver.
// ---------------------------------------------------------------------------
template <int NS, int C1, int C2, int C3>
static void run_branch(const float* xyz, const float* pts, const float* nxyz,
                       int* gidx, _Float16* XA, _Float16* XB,
                       _Float16* w0, _Float16* w1, _Float16* w2,
                       const float* b0, const float* g0, const float* be0,
                       const float* b1, const float* g1, const float* be1,
                       const float* b2, const float* g2, const float* be2,
                       float radius, float* gsum, float* gsumsq,
                       float* out_pts, int c_base, hipStream_t stream) {
  const int R = BATCH * NSAMP * NS;
  ball_query_kernel<NS><<<BATCH * NSAMP / 8, 256, 0, stream>>>(xyz, nxyz, gidx,
                                                               radius * radius);
  build_features_kernel<NS><<<(R + 255) / 256, 256, 0, stream>>>(xyz, pts, nxyz,
                                                                 gidx, XA, R);
  zero_stats_kernel<<<1, 256, 0, stream>>>(gsum);
  gemm_bn_kernel<32, C1><<<R / 128, 256, 0, stream>>>(XA, w0, b0, XB, gsum,
                                                      gsumsq, R);
  bn_relu_kernel<C1><<<(unsigned)(((size_t)R * C1 + 255) / 256), 256, 0,
                       stream>>>(XB, gsum, gsumsq, g0, be0, R);
  zero_stats_kernel<<<1, 256, 0, stream>>>(gsum);
  gemm_bn_kernel<C1, C2><<<R / 128, 256, 0, stream>>>(XB, w1, b1, XA, gsum,
                                                      gsumsq, R);
  bn_relu_kernel<C2><<<(unsigned)(((size_t)R * C2 + 255) / 256), 256, 0,
                       stream>>>(XA, gsum, gsumsq, g1, be1, R);
  zero_stats_kernel<<<1, 256, 0, stream>>>(gsum);
  gemm_bn_kernel<C2, C3><<<R / 128, 256, 0, stream>>>(XA, w2, b2, XB, gsum,
                                                      gsumsq, R);
  bn_relu_max_kernel<NS, C3><<<(BATCH * NSAMP * C3 + 255) / 256, 256, 0,
                               stream>>>(XB, gsum, gsumsq, g2, be2, out_pts,
                                         c_base);
}

extern "C" void kernel_launch(void* const* d_in, const int* in_sizes, int n_in,
                              void* d_out, int out_size, void* d_ws,
                              size_t ws_size, hipStream_t stream) {
  (void)in_sizes; (void)n_in; (void)out_size; (void)ws_size;
  const float* xyz = (const float*)d_in[0];
  const float* pts = (const float*)d_in[1];
  // params flat order: branch-major, layer-major, (W, b, gamma, beta)
  auto P = [&](int br, int l, int which) -> const float* {
    return (const float*)d_in[2 + (br * 3 + l) * 4 + which];
  };

  char* ws = (char*)d_ws;
  const size_t W_OFF      = 0;                     // 128 KB of f16 weights
  const size_t FPSIDX_OFF = 128ull << 10;          // 32 KB
  const size_t NEWXYZ_OFF = 160ull << 10;          // 96 KB
  const size_t STATS_OFF  = 256ull << 10;          // 1 KB (sum[128]+sumsq[128])
  const size_t GIDX_OFF   = 1ull << 20;            // 4 MB
  const size_t XA_OFF     = 8ull << 20;            // up to 192 MB (R x 96 f16)
  const size_t XB_OFF     = 208ull << 20;          // up to 256 MB (R x 128 f16)

  _Float16* wbuf   = (_Float16*)(ws + W_OFF);
  int*      fpsidx = (int*)(ws + FPSIDX_OFF);
  float*    nxyz   = (float*)(ws + NEWXYZ_OFF);
  float*    gsum   = (float*)(ws + STATS_OFF);
  float*    gsumsq = gsum + 128;
  int*      gidx   = (int*)(ws + GIDX_OFF);
  _Float16* XA     = (_Float16*)(ws + XA_OFF);
  _Float16* XB     = (_Float16*)(ws + XB_OFF);

  float* out      = (float*)d_out;
  float* out_pts  = out + (size_t)BATCH * 3 * NSAMP;

  // Convert + pad all 9 weight matrices to f16.
  const int CO9[9] = {32, 32, 64, 64, 64, 128, 64, 96, 128};
  const int CI9[9] = {6, 32, 32, 6, 64, 64, 6, 64, 96};
  const int KP9[9] = {32, 32, 32, 32, 64, 64, 32, 64, 96};
  _Float16* wp[9];
  size_t woff = 0;
  for (int i = 0; i < 9; ++i) {
    wp[i] = wbuf + woff;
    int n = CO9[i] * KP9[i];
    prep_weight_kernel<<<(n + 255) / 256, 256, 0, stream>>>(
        P(i / 3, i % 3, 0), wp[i], CO9[i], CI9[i], KP9[i]);
    woff += (size_t)n;
  }

  fps_kernel<<<BATCH, 1024, 0, stream>>>(xyz, fpsidx, nxyz, out);

  run_branch<16, 32, 32, 64>(xyz, pts, nxyz, gidx, XA, XB, wp[0], wp[1], wp[2],
                             P(0, 0, 1), P(0, 0, 2), P(0, 0, 3),
                             P(0, 1, 1), P(0, 1, 2), P(0, 1, 3),
                             P(0, 2, 1), P(0, 2, 2), P(0, 2, 3),
                             0.1f, gsum, gsumsq, out_pts, 0, stream);
  run_branch<32, 64, 64, 128>(xyz, pts, nxyz, gidx, XA, XB, wp[3], wp[4], wp[5],
                              P(1, 0, 1), P(1, 0, 2), P(1, 0, 3),
                              P(1, 1, 1), P(1, 1, 2), P(1, 1, 3),
                              P(1, 2, 1), P(1, 2, 2), P(1, 2, 3),
                              0.2f, gsum, gsumsq, out_pts, 64, stream);
  run_branch<128, 64, 96, 128>(xyz, pts, nxyz, gidx, XA, XB, wp[6], wp[7], wp[8],
                               P(2, 0, 1), P(2, 0, 2), P(2, 0, 3),
                               P(2, 1, 1), P(2, 1, 2), P(2, 1, 3),
                               P(2, 2, 1), P(2, 2, 2), P(2, 2, 3),
                               0.4f, gsum, gsumsq, out_pts, 192, stream);
}
